// PLEv3_3874060501468
// MI455X (gfx1250) — matrix operand order repaired and checked by reading, not compiled, not run on hardware
//
#include <hip/hip_runtime.h>
#include <hip/hip_bf16.h>
#include <math.h>

// ---------------------------------------------------------------------------
// Fully fused forward for the 6-expert MoE strategy model on gfx1250.
// All GEMMs run on v_wmma_f32_16x16x32_f16 (f16 in, f32 accum). Weights are
// repacked once per call into WMMA B-fragment tiles (f16) in d_ws by a small
// pack kernel. Each wave (32 lanes) owns a 16-row batch tile and carries it
// through the whole network using per-wave LDS staging (in-order DS ops, no
// barriers needed). Scalar epilogues (softmax/argmax/LN stats) use wave32
// shuffles. part_idx (d_in[2]) is an identity arange -> pure reshape, unused.
// ---------------------------------------------------------------------------

typedef _Float16 h16;
typedef __attribute__((ext_vector_type(16))) _Float16 v16h;
typedef __attribute__((ext_vector_type(8)))  _Float16 v8h;
typedef __attribute__((ext_vector_type(8)))  float    v8f;

#define EXPERTS 6
#define S_T 208            // stride (halfs) of X/Y staging tiles (>=192, mult of 8)
#define S_E 432            // stride (halfs) of EO tile (>=416, mult of 8)
#define WAVE_SMEM 28416    // bytes of LDS per wave

__device__ inline v8f vzero8() { v8f z = {0.f,0.f,0.f,0.f,0.f,0.f,0.f,0.f}; return z; }
__device__ inline float gelu_f(float x) { return 0.5f * x * (1.f + erff(x * 0.70710678118f)); }

// A-fragment (16x32 f16) from LDS row-major tile. ISA layout: lane L holds
// row m=L&15; lanes<16: V0..3 = K 0..7, V4..7 = K 16..23; lanes>=16: +8.
__device__ inline v16h load_a(const h16* lds, int S, int m, int g, int kcol) {
  const h16* base = lds + m * S + kcol + 8 * g;
  union { v16h v; v8h h[2]; } u;
  u.h[0] = *(const v8h*)(base);        // K = kcol+8g .. +7  -> V0..V3
  u.h[1] = *(const v8h*)(base + 16);   // K = kcol+16+8g ..  -> V4..V7
  return u.v;
}

// B-fragment: packed contiguously by pack_kernel, 512 halfs per 32x16 tile.
__device__ inline v16h load_b(const h16* pk, int tile, int lane) {
  return *(const v16h*)(pk + (size_t)tile * 512 + lane * 16);
}

__device__ inline v8f wmma_t(v16h a, v16h b, v8f c) {
  return __builtin_amdgcn_wmma_f32_16x16x32_f16(false, a, false, b, (short)0, c, false, false);
}

// Butterfly sum over the 16-lane half (wave32), per accumulator element.
__device__ inline void reduce16(float* v) {
#pragma unroll
  for (int r = 0; r < 8; ++r) {
    float t = v[r];
    t += __shfl_xor(t, 1); t += __shfl_xor(t, 2);
    t += __shfl_xor(t, 4); t += __shfl_xor(t, 8);
    v[r] = t;
  }
}

// C = gelu(A(16xK) @ Bpk + bias) -> f16 into LDS tile C at column ccol.
__device__ inline void gemm_gelu_store(const h16* A, int As, const h16* Bpk,
                                       const float* bias, h16* C, int Cs, int ccol,
                                       int Kc, int Nc, int lane, int m, int g) {
  const int m_hi = g * 8, n_l = lane & 15;
  for (int nc = 0; nc < Nc; ++nc) {
    v8f c = vzero8();
    for (int kc = 0; kc < Kc; ++kc)
      c = wmma_t(load_a(A, As, m, g, kc * 32), load_b(Bpk, kc * Nc + nc, lane), c);
    float bb = bias[nc * 16 + n_l];
#pragma unroll
    for (int r = 0; r < 8; ++r)
      C[(r + m_hi) * Cs + ccol + nc * 16 + n_l] = (h16)gelu_f(c[r] + bb);
  }
}

// Small head projection 64(or 32)->n logits (n<=16, N padded to 16 in pack).
__device__ inline void head_logits(const h16* A, int As, const h16* Bpk,
                                   const float* b2, int nvalid, int Kc,
                                   float* SCf, int lane, int m, int g) {
  const int n_l = lane & 15;
  v8f c = vzero8();
  for (int kc = 0; kc < Kc; ++kc)
    c = wmma_t(load_a(A, As, m, g, kc * 32), load_b(Bpk, kc, lane), c);
  if (n_l < nvalid) {
    float bb = b2[n_l];
#pragma unroll
    for (int r = 0; r < 8; ++r) SCf[(r + g * 8) * 16 + n_l] = c[r] + bb;
  }
}

// Scalar per-row: logits -> write logits+softmax, return argmax.
__device__ inline int head_scalar(const float* SCf, int row, int n,
                                  float* o_l, float* o_p, size_t gidx) {
  float l[5], e[5], mx = -1e30f, se = 0.f;
  for (int j = 0; j < n; ++j) { l[j] = SCf[row * 16 + j]; mx = fmaxf(mx, l[j]); }
  for (int j = 0; j < n; ++j) { e[j] = __expf(l[j] - mx); se += e[j]; }
  int am = 0; float bm = e[0];
  for (int j = 1; j < n; ++j) if (e[j] > bm) { bm = e[j]; am = j; }
  float inv = 1.f / se;
  for (int j = 0; j < n; ++j) { o_l[gidx * n + j] = l[j]; o_p[gidx * n + j] = e[j] * inv; }
  return am;
}

// ---------------------------------------------------------------------------
// Weight repack: f32 (Ksrc x Nsrc row-major) -> f16 WMMA B-fragment tiles of
// padded (Kp x Np). Tile t = kc*(Np/16)+nc, element [lane][j]: per ISA B layout
// lanes 0-15 hold K 0..7,16..23 of the 32-chunk, lanes 16-31 hold K 8..15,24..31;
// element j<8 -> K 8h+j, j>=8 -> K 16+8h+(j-8); column = lane&15.
// ---------------------------------------------------------------------------
__global__ void pack_kernel(const float* __restrict__ W, int Kp, int Np,
                            int Ksrc, int Nsrc, h16* __restrict__ dst) {
  int idx = blockIdx.x * 256 + threadIdx.x;
  int tiles = (Kp / 32) * (Np / 16);
  if (idx >= tiles * 512) return;
  int t = idx >> 9, r = idx & 511;
  int lane = r >> 4, j = r & 15;
  int Nc = Np / 16;
  int kc = t / Nc, nc = t % Nc;
  int v = j >> 1, pbit = j & 1;
  int kl = ((lane >= 16) ? 8 : 0) + ((2 * v + pbit) & 7) + (v >= 4 ? 16 : 0);
  int nl = lane & 15;
  int k = kc * 32 + kl, n = nc * 16 + nl;
  float val = (k < Ksrc && n < Nsrc) ? W[(size_t)k * Nsrc + n] : 0.f;
  dst[idx] = (h16)val;
}

struct KParams {
  const float *features, *account;
  const h16 *w1, *w2, *w3, *wq, *wk, *wf, *rg1, *rg2, *tf1, *tf2, *rr1, *rr2,
            *mae1, *mae2, *mfe1, *mfe2, *cf1, *cf2;
  const float *b1, *b2, *b3, *lng, *lnb, *Wa, *ba, *bf, *flng, *flnb, *bq, *bk;
  const float *rgb1, *rgb2, *remb, *tfb1, *tfb2, *temb, *rrb1, *rrb2;
  const float *maeb1, *maeb2, *mfeb1, *mfeb2, *cfb1, *cfb2;
  float* out; int B;
};

__global__ void __launch_bounds__(128) fused_forward(KParams p) {
  extern __shared__ char smem[];
  const int tid = threadIdx.x;
  const int wave = tid >> 5, lane = tid & 31;
  const int m = lane & 15, g = lane >> 4, m_hi = g * 8, n_l = m;

  char* wb = smem + wave * WAVE_SMEM;
  h16* X  = (h16*)wb;            // 16 x S_T staging
  h16* Y  = X + 16 * S_T;        // 16 x S_T staging
  h16* EO = Y + 16 * S_T;        // 16 x S_E : eo_flat(384) | acc(32)
  float* SCf = (float*)(EO + 16 * S_E);  // 16 x 16 floats (scores/logits)
  int*   SCi = (int*)(SCf + 256);        // 16 x {ri,ti,rri}

  const int row0 = (blockIdx.x * 4 + wave) * 16;
  if (row0 >= p.B) return;
  const size_t B_ = (size_t)p.B;
  float* o_rl  = p.out;
  float* o_tl  = o_rl  + B_ * 4;
  float* o_rrl = o_tl  + B_ * 5;
  float* o_rp  = o_rrl + B_ * 5;
  float* o_tp  = o_rp  + B_ * 4;
  float* o_rrp = o_tp  + B_ * 5;
  float* o_si  = o_rrp + B_ * 5;
  float* o_mae = o_si  + B_;
  float* o_mfe = o_mae + B_;
  float* o_cf  = o_mfe + B_;
  float* o_gw  = o_cf  + B_;

  // ---------------- experts ----------------
  const float* frow = p.features + (size_t)(row0 + m) * 192;
  for (int e = 0; e < EXPERTS; ++e) {
    // A fragment of xs (16x32) straight from global f32, converted to f16.
    v16h a;
#pragma unroll
    for (int v = 0; v < 8; ++v) {
      int k = (v < 4 ? 2 * v : 2 * v + 8) + 8 * g;
      a[2 * v]     = (h16)frow[e * 32 + k];
      a[2 * v + 1] = (h16)frow[e * 32 + k + 1];
    }
    // h1 = gelu(xs @ W1[e] + b1[e])  (32 -> 128) -> X
    const h16* w1e = p.w1 + (size_t)e * 8 * 512;
    const float* b1e = p.b1 + e * 128;
    for (int nc = 0; nc < 8; ++nc) {
      v8f c = wmma_t(a, load_b(w1e, nc, lane), vzero8());
      float bb = b1e[nc * 16 + n_l];
#pragma unroll
      for (int r = 0; r < 8; ++r)
        X[(r + m_hi) * S_T + nc * 16 + n_l] = (h16)gelu_f(c[r] + bb);
    }
    // h2 = gelu(h1 @ W2[e] + b2[e])  (128 -> 128) -> Y
    gemm_gelu_store(X, S_T, p.w2 + (size_t)e * 32 * 512, p.b2 + e * 128,
                    Y, S_T, 0, 4, 8, lane, m, g);
    // eo = LN(h2 @ W3[e] + b3[e])  (128 -> 64) -> EO[:, e*64:]
    {
      const h16* w3e = p.w3 + (size_t)e * 16 * 512;
      const float* b3e = p.b3 + e * 64;
      const float* ge = p.lng + e * 64;
      const float* be = p.lnb + e * 64;
      v8f c[4];
      for (int nc = 0; nc < 4; ++nc) {
        c[nc] = vzero8();
        for (int kc = 0; kc < 4; ++kc)
          c[nc] = wmma_t(load_a(Y, S_T, m, g, kc * 32), load_b(w3e, kc * 4 + nc, lane), c[nc]);
        float bb = b3e[nc * 16 + n_l];
#pragma unroll
        for (int r = 0; r < 8; ++r) c[nc][r] += bb;
      }
      float s[8], ss[8];
#pragma unroll
      for (int r = 0; r < 8; ++r) {
        float a0 = c[0][r], a1 = c[1][r], a2 = c[2][r], a3 = c[3][r];
        s[r] = a0 + a1 + a2 + a3;
        ss[r] = a0 * a0 + a1 * a1 + a2 * a2 + a3 * a3;
      }
      reduce16(s); reduce16(ss);
      for (int nc = 0; nc < 4; ++nc) {
        float gg = ge[nc * 16 + n_l], bv = be[nc * 16 + n_l];
#pragma unroll
        for (int r = 0; r < 8; ++r) {
          float mean = s[r] * (1.f / 64.f);
          float var  = ss[r] * (1.f / 64.f) - mean * mean;
          float val = (c[nc][r] - mean) * rsqrtf(var + 1e-5f) * gg + bv;
          EO[(r + m_hi) * S_E + e * 64 + nc * 16 + n_l] = (h16)val;
        }
      }
    }
  }

  // acc = gelu(account @ Wa + ba) (4 -> 32), one column per lane -> EO[:,384:]
  {
    float wcol[4];
#pragma unroll
    for (int j = 0; j < 4; ++j) wcol[j] = p.Wa[j * 32 + lane];
    float bb = p.ba[lane];
    for (int mm = 0; mm < 16; ++mm) {
      const float* ar = p.account + (size_t)(row0 + mm) * 4;
      float s2 = bb;
#pragma unroll
      for (int j = 0; j < 4; ++j) s2 += ar[j] * wcol[j];
      EO[mm * S_E + 384 + lane] = (h16)gelu_f(s2);
    }
  }

  // q = [eo_flat | acc] @ Wq + bq  (416 -> 64), keep fragments in VGPRs
  v8f qf[4];
  for (int nc = 0; nc < 4; ++nc) {
    qf[nc] = vzero8();
    for (int kc = 0; kc < 13; ++kc)
      qf[nc] = wmma_t(load_a(EO, S_E, m, g, kc * 32), load_b(p.wq, kc * 4 + nc, lane), qf[nc]);
    float bb = p.bq[nc * 16 + n_l];
#pragma unroll
    for (int r = 0; r < 8; ++r) qf[nc][r] += bb;
  }

  // scores[e] = (eo_e @ Wk[e] + bk[e]) . q / sqrt(64)
  for (int e = 0; e < EXPERTS; ++e) {
    const h16* wke = p.wk + (size_t)e * 8 * 512;
    const float* bke = p.bk + e * 64;
    float d[8];
#pragma unroll
    for (int r = 0; r < 8; ++r) d[r] = 0.f;
    for (int nc = 0; nc < 4; ++nc) {
      v8f kf = vzero8();
      for (int kk = 0; kk < 2; ++kk)
        kf = wmma_t(load_a(EO, S_E, m, g, e * 64 + kk * 32), load_b(wke, kk * 4 + nc, lane), kf);
      float bb = bke[nc * 16 + n_l];
#pragma unroll
      for (int r = 0; r < 8; ++r) d[r] += (kf[r] + bb) * qf[nc][r];
    }
    reduce16(d);
    if (n_l == 0) {
#pragma unroll
      for (int r = 0; r < 8; ++r) SCf[(r + m_hi) * 16 + e] = d[r] * 0.125f;
    }
  }

  // gating softmax over E=6; write gw
  if (lane < 16) {
    float sc[6], mx = -1e30f, se = 0.f;
    for (int e = 0; e < 6; ++e) { sc[e] = SCf[lane * 16 + e]; mx = fmaxf(mx, sc[e]); }
    for (int e = 0; e < 6; ++e) { sc[e] = __expf(sc[e] - mx); se += sc[e]; }
    float inv = 1.f / se;
    for (int e = 0; e < 6; ++e) {
      float w = sc[e] * inv;
      SCf[lane * 16 + e] = w;
      o_gw[(size_t)(row0 + lane) * 6 + e] = w;
    }
  }

  // gated = sum_e gw[e]*eo_e ; build [gated | acc] (16x96) -> X
  for (int mm = 0; mm < 16; ++mm) {
    float w[6];
#pragma unroll
    for (int e = 0; e < 6; ++e) w[e] = SCf[mm * 16 + e];
    int c0 = lane * 2;
    float g0 = 0.f, g1 = 0.f;
#pragma unroll
    for (int e = 0; e < 6; ++e) {
      g0 += w[e] * (float)EO[mm * S_E + e * 64 + c0];
      g1 += w[e] * (float)EO[mm * S_E + e * 64 + c0 + 1];
    }
    X[mm * S_T + c0]     = (h16)g0;
    X[mm * S_T + c0 + 1] = (h16)g1;
    X[mm * S_T + 64 + lane] = EO[mm * S_E + 384 + lane];
  }

  // fused = LN(gelu(X(16x96) @ Wf + bf))  (96 -> 128) -> Y[:, :128]
  {
    v8f c[8];
    for (int nc = 0; nc < 8; ++nc) {
      c[nc] = vzero8();
      for (int kc = 0; kc < 3; ++kc)
        c[nc] = wmma_t(load_a(X, S_T, m, g, kc * 32), load_b(p.wf, kc * 8 + nc, lane), c[nc]);
      float bb = p.bf[nc * 16 + n_l];
#pragma unroll
      for (int r = 0; r < 8; ++r) c[nc][r] = gelu_f(c[nc][r] + bb);
    }
    float s[8], ss[8];
#pragma unroll
    for (int r = 0; r < 8; ++r) { s[r] = 0.f; ss[r] = 0.f; }
    for (int nc = 0; nc < 8; ++nc) {
#pragma unroll
      for (int r = 0; r < 8; ++r) { float x = c[nc][r]; s[r] += x; ss[r] += x * x; }
    }
    reduce16(s); reduce16(ss);
    for (int nc = 0; nc < 8; ++nc) {
      float gg = p.flng[nc * 16 + n_l], bv = p.flnb[nc * 16 + n_l];
#pragma unroll
      for (int r = 0; r < 8; ++r) {
        float mean = s[r] * (1.f / 128.f);
        float var  = ss[r] * (1.f / 128.f) - mean * mean;
        Y[(r + m_hi) * S_T + nc * 16 + n_l] =
            (h16)((c[nc][r] - mean) * rsqrtf(var + 1e-5f) * gg + bv);
      }
    }
  }

  // regime head: fused(128) -> 64 -> 4 ; remb -> Y[:,128:160]
  gemm_gelu_store(Y, S_T, p.rg1, p.rgb1, X, S_T, 0, 4, 4, lane, m, g);
  head_logits(X, S_T, p.rg2, p.rgb2, 4, 2, SCf, lane, m, g);
  if (lane < 16) {
    int row = lane;
    int am = head_scalar(SCf, row, 4, o_rl, o_rp, (size_t)(row0 + row));
    SCi[row] = am;
    for (int j = 0; j < 32; ++j) Y[row * S_T + 128 + j] = (h16)p.remb[am * 32 + j];
  }

  // timeframe head: [fused|remb](160) -> 64 -> 5 ; temb -> Y[:,160:192]
  gemm_gelu_store(Y, S_T, p.tf1, p.tfb1, X, S_T, 0, 5, 4, lane, m, g);
  head_logits(X, S_T, p.tf2, p.tfb2, 5, 2, SCf, lane, m, g);
  if (lane < 16) {
    int row = lane;
    int am = head_scalar(SCf, row, 5, o_tl, o_tp, (size_t)(row0 + row));
    SCi[16 + row] = am;
    for (int j = 0; j < 32; ++j) Y[row * S_T + 160 + j] = (h16)p.temb[am * 32 + j];
  }

  // risk head: [fused|remb|temb](192) -> 64 -> 5 ; strategy index
  gemm_gelu_store(Y, S_T, p.rr1, p.rrb1, X, S_T, 0, 6, 4, lane, m, g);
  head_logits(X, S_T, p.rr2, p.rrb2, 5, 2, SCf, lane, m, g);
  if (lane < 16) {
    int row = lane;
    int am = head_scalar(SCf, row, 5, o_rrl, o_rrp, (size_t)(row0 + row));
    SCi[32 + row] = am;
    o_si[row0 + row] = (float)(SCi[row] * 25 + SCi[16 + row] * 5 + am);
  }

  // pred_in = [fused(128) | onehot4 | onehot5 | onehot5 | zero-pad to 160] -> X
  for (int mm = 0; mm < 16; ++mm)
    for (int cc = lane; cc < 128; cc += 32) X[mm * S_T + cc] = Y[mm * S_T + cc];
  if (lane < 16) {
    int row = lane;
    for (int j = 128; j < 160; ++j) X[row * S_T + j] = (h16)0.f;
    X[row * S_T + 128 + SCi[row]]      = (h16)1.f;
    X[row * S_T + 132 + SCi[16 + row]] = (h16)1.f;
    X[row * S_T + 137 + SCi[32 + row]] = (h16)1.f;
  }

  // mae head: pred_in(160 padded) -> 64 -> 1
  gemm_gelu_store(X, S_T, p.mae1, p.maeb1, EO, S_E, 0, 5, 4, lane, m, g);
  {
    v8f c = vzero8();
    for (int kc = 0; kc < 2; ++kc)
      c = wmma_t(load_a(EO, S_E, m, g, kc * 32), load_b(p.mae2, kc, lane), c);
    if (n_l == 0) {
      float bb = p.maeb2[0];
#pragma unroll
      for (int r = 0; r < 8; ++r) o_mae[row0 + r + m_hi] = c[r] + bb;
    }
  }
  // mfe head
  gemm_gelu_store(X, S_T, p.mfe1, p.mfeb1, EO, S_E, 0, 5, 4, lane, m, g);
  {
    v8f c = vzero8();
    for (int kc = 0; kc < 2; ++kc)
      c = wmma_t(load_a(EO, S_E, m, g, kc * 32), load_b(p.mfe2, kc, lane), c);
    if (n_l == 0) {
      float bb = p.mfeb2[0];
#pragma unroll
      for (int r = 0; r < 8; ++r) o_mfe[row0 + r + m_hi] = c[r] + bb;
    }
  }
  // confidence head: fused(128) -> 32 -> 1, sigmoid
  gemm_gelu_store(Y, S_T, p.cf1, p.cfb1, EO, S_E, 0, 4, 2, lane, m, g);
  {
    v8f c = wmma_t(load_a(EO, S_E, m, g, 0), load_b(p.cf2, 0, lane), vzero8());
    if (n_l == 0) {
      float bb = p.cfb2[0];
#pragma unroll
      for (int r = 0; r < 8; ++r) {
        float x = c[r] + bb;
        o_cf[row0 + r + m_hi] = 1.f / (1.f + __expf(-x));
      }
    }
  }
}

extern "C" void kernel_launch(void* const* d_in, const int* in_sizes, int n_in,
                              void* d_out, int out_size, void* d_ws, size_t ws_size,
                              hipStream_t stream) {
  (void)n_in; (void)out_size; (void)ws_size;
  const float* features = (const float*)d_in[0];
  const float* account  = (const float*)d_in[1];
  // d_in[2] = part_idx (identity) unused
  const float* W1  = (const float*)d_in[3];  const float* b1   = (const float*)d_in[4];
  const float* W2  = (const float*)d_in[5];  const float* b2   = (const float*)d_in[6];
  const float* W3  = (const float*)d_in[7];  const float* b3   = (const float*)d_in[8];
  const float* lng = (const float*)d_in[9];  const float* lnb  = (const float*)d_in[10];
  const float* Wa  = (const float*)d_in[11]; const float* ba   = (const float*)d_in[12];
  const float* Wf  = (const float*)d_in[13]; const float* bf   = (const float*)d_in[14];
  const float* flng= (const float*)d_in[15]; const float* flnb = (const float*)d_in[16];
  const float* Wq  = (const float*)d_in[17]; const float* bq   = (const float*)d_in[18];
  const float* Wk  = (const float*)d_in[19]; const float* bk   = (const float*)d_in[20];
  const float* rg1 = (const float*)d_in[21]; const float* rgb1 = (const float*)d_in[22];
  const float* rg2 = (const float*)d_in[23]; const float* rgb2 = (const float*)d_in[24];
  const float* remb= (const float*)d_in[25];
  const float* tf1 = (const float*)d_in[26]; const float* tfb1 = (const float*)d_in[27];
  const float* tf2 = (const float*)d_in[28]; const float* tfb2 = (const float*)d_in[29];
  const float* temb= (const float*)d_in[30];
  const float* rr1 = (const float*)d_in[31]; const float* rrb1 = (const float*)d_in[32];
  const float* rr2 = (const float*)d_in[33]; const float* rrb2 = (const float*)d_in[34];
  const float* mae1= (const float*)d_in[35]; const float* maeb1= (const float*)d_in[36];
  const float* mae2= (const float*)d_in[37]; const float* maeb2= (const float*)d_in[38];
  const float* mfe1= (const float*)d_in[39]; const float* mfeb1= (const float*)d_in[40];
  const float* mfe2= (const float*)d_in[41]; const float* mfeb2= (const float*)d_in[42];
  const float* cf1 = (const float*)d_in[43]; const float* cfb1 = (const float*)d_in[44];
  const float* cf2 = (const float*)d_in[45]; const float* cfb2 = (const float*)d_in[46];

  const int B = in_sizes[0] / 192;
  h16* ws = (h16*)d_ws;
  size_t off = 0;   // total packed weights ~296k halfs (~0.6 MB) in d_ws

  auto pack1 = [&](const float* W, int Kp, int Np, int Ks, int Ns) -> const h16* {
    h16* dst = ws + off;
    size_t total = (size_t)(Kp / 32) * (Np / 16) * 512;
    off += total;
    pack_kernel<<<dim3((unsigned)((total + 255) / 256)), dim3(256), 0, stream>>>(
        W, Kp, Np, Ks, Ns, dst);
    return dst;
  };
  auto packE = [&](const float* W, int Kp, int Np) -> const h16* {
    const h16* base = ws + off;
    for (int e = 0; e < EXPERTS; ++e) pack1(W + (size_t)e * Kp * Np, Kp, Np, Kp, Np);
    return base;
  };

  KParams p{};
  p.w1  = packE(W1, 32, 128);
  p.w2  = packE(W2, 128, 128);
  p.w3  = packE(W3, 128, 64);
  p.wq  = pack1(Wq, 416, 64, 416, 64);
  p.wk  = packE(Wk, 64, 64);
  p.wf  = pack1(Wf, 96, 128, 96, 128);
  p.rg1 = pack1(rg1, 128, 64, 128, 64);
  p.rg2 = pack1(rg2, 64, 16, 64, 4);     // N padded 4 -> 16
  p.tf1 = pack1(tf1, 160, 64, 160, 64);
  p.tf2 = pack1(tf2, 64, 16, 64, 5);     // N padded 5 -> 16
  p.rr1 = pack1(rr1, 192, 64, 192, 64);
  p.rr2 = pack1(rr2, 64, 16, 64, 5);
  p.mae1= pack1(mae1, 160, 64, 142, 64); // K padded 142 -> 160
  p.mae2= pack1(mae2, 64, 16, 64, 1);
  p.mfe1= pack1(mfe1, 160, 64, 142, 64);
  p.mfe2= pack1(mfe2, 64, 16, 64, 1);
  p.cf1 = pack1(cf1, 128, 32, 128, 32);
  p.cf2 = pack1(cf2, 32, 16, 32, 1);

  p.features = features; p.account = account;
  p.b1 = b1; p.b2 = b2; p.b3 = b3; p.lng = lng; p.lnb = lnb;
  p.Wa = Wa; p.ba = ba; p.bf = bf; p.flng = flng; p.flnb = flnb;
  p.bq = bq; p.bk = bk;
  p.rgb1 = rgb1; p.rgb2 = rgb2; p.remb = remb;
  p.tfb1 = tfb1; p.tfb2 = tfb2; p.temb = temb;
  p.rrb1 = rrb1; p.rrb2 = rrb2;
  p.maeb1 = maeb1; p.maeb2 = maeb2; p.mfeb1 = mfeb1; p.mfeb2 = mfeb2;
  p.cfb1 = cfb1; p.cfb2 = cfb2;
  p.out = (float*)d_out; p.B = B;

  const int grid = (B + 63) / 64;   // 4 waves x 16 rows per block
  fused_forward<<<dim3(grid), dim3(128), 4 * WAVE_SMEM, stream>>>(p);
}
